// FLDEncoder_13941463843384
// MI455X (gfx1250) — matrix operand order repaired and checked by reading, not compiled, not run on hardware
//
#include <hip/hip_runtime.h>
#include <hip/hip_bf16.h>
#include <cstddef>

// ---------------------------------------------------------------------------
// FLDEncoder on MI455X (gfx1250, wave32, WMMA 16x16x32 f16->f32)
//
//   conv1: implicit GEMM  M=H=256, N=B*T=16320, K=IN*256(tau pad)=8192
//   conv2: implicit GEMM  M=E=128, N=16320,     K=H*256(tau pad)=65536
//   mlp1 : per-e GEMM     M=H=256, N=B=64,      K=256 (t pad)
//
// Block tile: 64 M (4 waves x 16) x 64 N (4 sub-tiles x 16), so each A
// fragment feeds 4 WMMAs -> 4x less weight traffic from L2 vs a 16-wide tile.
// B-panels staged in LDS in WMMA-fragment order (one v16h load per fragment).
//
// Spectral block via Parseval (odd N=255):
//   sum_{f=0..127} |rfft|^2 = (N*sum z^2 + (sum z)^2)/2
// frequency[b,e] == freq_bins[e] exactly (reference broadcasts freq_bins over
// the channel axis, so the power ratio cancels).
// ---------------------------------------------------------------------------

typedef __attribute__((ext_vector_type(16))) _Float16 v16h;
typedef __attribute__((ext_vector_type(8)))  _Float16 v8h;
typedef __attribute__((ext_vector_type(8)))  float    v8f;

#define NB   64
#define NT   255
#define NIN  32
#define NH   256
#define NE   128
#define NCOL (NB * NT)          // 16320
#define K1   8192               // IN * 256 (tau padded 255->256)
#define K2   65536              // H * 256  (tau padded)

// ---------------------- workspace layout (bytes) ---------------------------
static constexpr size_t OFF_A1  = 0;                         // 256*8192*2   = 4,194,304
static constexpr size_t OFF_A2  = OFF_A1 + 4194304;          // 128*65536*2  = 16,777,216
static constexpr size_t OFF_X16 = OFF_A2 + 16777216;         // 64*255*32*2  -> pad 1,048,576
static constexpr size_t OFF_P1  = OFF_X16 + 1048576;         // 128*256*256*2 = 16,777,216
static constexpr size_t OFF_H   = OFF_P1 + 16777216;         // 64*255*256*4 -> pad 16,777,216
static constexpr size_t OFF_HN  = OFF_H + 16777216;          // f16           -> pad 8,388,608
static constexpr size_t OFF_Z   = OFF_HN + 8388608;          // 64*255*128*4 -> pad 8,388,608
static constexpr size_t OFF_HM  = OFF_Z + 8388608;           // 128*64*256*4  = 8,388,608
static constexpr size_t OFF_ST  = OFF_HM + 8388608;          // 1024 floats (sum,sumsq,scale,shift)

// ---------------------- cast / pack kernels --------------------------------

__global__ __launch_bounds__(256) void cast_a1(const float* __restrict__ w,
                                               _Float16* __restrict__ A1,
                                               float* __restrict__ stats) {
  int idx = blockIdx.x * 256 + threadIdx.x;          // 256*8192 total
  if (idx < 512) stats[idx] = 0.0f;
  int c = idx >> 13, k = idx & 8191;
  int tau = k >> 5, i = k & 31;
  float v = (tau < NT) ? w[(c * NIN + i) * NT + tau] : 0.0f;
  A1[idx] = (_Float16)v;
}

__global__ __launch_bounds__(256) void cast_a2(const float* __restrict__ w,
                                               _Float16* __restrict__ A2) {
  int idx = blockIdx.x * 256 + threadIdx.x;          // 128*65536 total
  int e = idx >> 16, k = idx & 65535;
  int tau = k >> 8, c = k & 255;
  float v = (tau < NT) ? w[(e * NH + c) * NT + tau] : 0.0f;
  A2[idx] = (_Float16)v;
}

__global__ __launch_bounds__(256) void cast_x(const float* __restrict__ s,
                                              _Float16* __restrict__ X) {
  int idx = blockIdx.x * 256 + threadIdx.x;          // 64*255*32 = 522240
  X[idx] = (_Float16)s[idx];
}

__global__ __launch_bounds__(256) void cast_p1(const float* __restrict__ w,
                                               _Float16* __restrict__ P1) {
  int idx = blockIdx.x * 256 + threadIdx.x;          // 128*256*256 total
  int e = idx >> 16, h = (idx >> 8) & 255, t = idx & 255;
  float v = (t < NT) ? w[(e * NH + h) * NT + t] : 0.0f;
  P1[idx] = (_Float16)v;
}

// ------------------- WMMA fragment helpers ---------------------------------
// A (16x32 f16): lane row m=lane&15; element j -> k = 16*hl + j
//   => two contiguous 8-half runs at k0+16*hl and k0+16*hl+8.
// B (32x16 f16): lane col n=lane&15; element j -> k = 16*hl + j.
//   LDS sub-panel (8KB per N-sub-tile) stored fragment-ordered:
//     addr(kk,nn) = sub*4096 + ((kk>>5)*32 + ((kk>>4)&1)*16 + nn)*16 + (kk&15)
// C/D (16x16 f32): element r -> row 8*hl + r, col lane&15.

__device__ __forceinline__ v16h load_a_frag(const _Float16* __restrict__ arow,
                                            int k0, int hl) {
  v8h a0 = *(const v8h*)(arow + k0 + (hl << 4));
  v8h a1 = *(const v8h*)(arow + k0 + 8 + (hl << 4));
  return __builtin_shufflevector(a0, a1, 0, 1, 2, 3, 4, 5, 6, 7,
                                 8, 9, 10, 11, 12, 13, 14, 15);
}

__device__ __forceinline__ void store_b_run(_Float16* dst, const v8h* src) {
  v8h x0 = src[0], x1 = src[1], x2 = src[2], x3 = src[3];
  *(v8h*)(dst) = x0;        *(v8h*)(dst + 8) = x1;        // k' = 0..15
  *(v8h*)(dst + 256) = x2;  *(v8h*)(dst + 264) = x3;      // k' = 16..31
}

__device__ __forceinline__ void store_b_zero(_Float16* dst) {
  v8h z = {};
  *(v8h*)(dst) = z;       *(v8h*)(dst + 8) = z;
  *(v8h*)(dst + 256) = z; *(v8h*)(dst + 264) = z;
}

// ---------------------- conv1 implicit GEMM --------------------------------
// grid (4, 255), block 128 (4 waves). Block tile: 64 M x 64 N.
__global__ __launch_bounds__(128) void conv1_gemm(const _Float16* __restrict__ A1,
                                                  const _Float16* __restrict__ X16,
                                                  const float* __restrict__ bias,
                                                  float* __restrict__ hbuf,
                                                  float* __restrict__ stats) {
  __shared__ __align__(32) _Float16 sB[16384];       // 32 KB: 4 sub-panels
  const int lane = threadIdx.x & 31;
  const int hl = lane >> 4, lm = lane & 15;
  const int mBase = blockIdx.x * 64 + (threadIdx.x >> 5) * 16;
  const int nBase = blockIdx.y * 64;

  // staging role: thread owns (k-chunk g, column nn) across the 4 sub-tiles
  const int g = threadIdx.x >> 4;                    // 0..7 (tau within chunk)
  const int nn = threadIdx.x & 15;
  int bsv[4], tsv[4];
#pragma unroll
  for (int sub = 0; sub < 4; ++sub) {
    int ns = nBase + sub * 16 + nn;
    bsv[sub] = ns / NT;
    tsv[sub] = ns - bsv[sub] * NT;
  }
  _Float16* dst0 = sB + g * 512 + nn * 16;

  v8f acc[4] = {};
  const _Float16* __restrict__ arow = A1 + (size_t)(mBase + lm) * K1;

  for (int kc = 0; kc < 32; ++kc) {                  // 32 outer * 256 k each
    const int tau = kc * 8 + g;
#pragma unroll
    for (int sub = 0; sub < 4; ++sub) {
      int tt = tsv[sub] + tau - 127;
      _Float16* d = dst0 + sub * 4096;
      if (tt >= 0 && tt < NT)
        store_b_run(d, (const v8h*)(X16 + ((size_t)bsv[sub] * NT + tt) * NIN));
      else
        store_b_zero(d);
    }
    __syncthreads();
    __builtin_prefetch(arow + (size_t)(kc + 1) * 256, 0, 3);  // global_prefetch_b8
#pragma unroll
    for (int s = 0; s < 8; ++s) {
      v16h af = load_a_frag(arow, kc * 256 + s * 32, hl);
#pragma unroll
      for (int sub = 0; sub < 4; ++sub) {
        v16h bf = *(const v16h*)(sB + sub * 4096 + s * 512 + hl * 256 + lm * 16);
        acc[sub] = __builtin_amdgcn_wmma_f32_16x16x32_f16(false, af, false, bf,
                                                          (short)0, acc[sub],
                                                          false, false);
      }
    }
    __syncthreads();
  }

  // epilogue: bias + ReLU, vector stores, BN sum/sumsq (subs share channels)
  const int c0 = mBase + hl * 8;
  float biasv[8];
#pragma unroll
  for (int r = 0; r < 8; ++r) biasv[r] = bias[c0 + r];

  v8f outv[4];
#pragma unroll
  for (int sub = 0; sub < 4; ++sub) {
    const int n = nBase + sub * 16 + lm;
    const int b = n / NT, t = n - b * NT;
#pragma unroll
    for (int r = 0; r < 8; ++r) {
      float v = acc[sub][r] + biasv[r];
      outv[sub][r] = v > 0.0f ? v : 0.0f;
    }
    *(v8f*)(hbuf + ((size_t)b * NT + t) * NH + c0) = outv[sub];
  }
#pragma unroll
  for (int r = 0; r < 8; ++r) {
    float s1 = outv[0][r] + outv[1][r] + outv[2][r] + outv[3][r];
    float s2 = outv[0][r] * outv[0][r] + outv[1][r] * outv[1][r] +
               outv[2][r] * outv[2][r] + outv[3][r] * outv[3][r];
#pragma unroll
    for (int m = 8; m >= 1; m >>= 1) {               // reduce over 16 n-lanes
      s1 += __shfl_xor(s1, m, 32);
      s2 += __shfl_xor(s2, m, 32);
    }
    if (lm == 0) {
      atomicAdd(&stats[c0 + r], s1);
      atomicAdd(&stats[NH + c0 + r], s2);
    }
  }
}

// ---------------------- BatchNorm finalize / apply -------------------------
__global__ __launch_bounds__(256) void bn_finalize(const float* __restrict__ gamma,
                                                   const float* __restrict__ beta,
                                                   float* __restrict__ stats) {
  int c = threadIdx.x;
  const float invN = 1.0f / (float)NCOL;
  float mean = stats[c] * invN;
  float var = stats[NH + c] * invN - mean * mean;
  float rstd = rsqrtf(var + 1e-5f);
  float sc = gamma[c] * rstd;
  stats[512 + c] = sc;
  stats[768 + c] = beta[c] - mean * sc;
}

__global__ __launch_bounds__(256) void bn_apply(const float* __restrict__ hbuf,
                                                const float* __restrict__ stats,
                                                _Float16* __restrict__ hn) {
  int idx = blockIdx.x * 256 + threadIdx.x;          // 64*255*256 total
  int c = idx & 255;
  hn[idx] = (_Float16)(hbuf[idx] * stats[512 + c] + stats[768 + c]);
}

// ---------------------- conv2 implicit GEMM --------------------------------
// grid (2, 255), block 128. Block tile 64 M x 64 N, K2 = 65536.
__global__ __launch_bounds__(128) void conv2_gemm(const _Float16* __restrict__ A2,
                                                  const _Float16* __restrict__ HN,
                                                  const float* __restrict__ bias,
                                                  float* __restrict__ zbuf) {
  __shared__ __align__(32) _Float16 sB[16384];
  const int lane = threadIdx.x & 31;
  const int hl = lane >> 4, lm = lane & 15;
  const int mBase = blockIdx.x * 64 + (threadIdx.x >> 5) * 16;
  const int nBase = blockIdx.y * 64;

  const int g = threadIdx.x >> 4;                    // channel chunk: c = g*32..+31
  const int nn = threadIdx.x & 15;
  int bsv[4], tsv[4];
#pragma unroll
  for (int sub = 0; sub < 4; ++sub) {
    int ns = nBase + sub * 16 + nn;
    bsv[sub] = ns / NT;
    tsv[sub] = ns - bsv[sub] * NT;
  }
  _Float16* dst0 = sB + g * 512 + nn * 16;

  v8f acc[4] = {};
  const _Float16* __restrict__ arow = A2 + (size_t)(mBase + lm) * K2;

  for (int kc = 0; kc < 256; ++kc) {                 // kc == tau
#pragma unroll
    for (int sub = 0; sub < 4; ++sub) {
      int tt = tsv[sub] + kc - 127;
      _Float16* d = dst0 + sub * 4096;
      if (kc < NT && tt >= 0 && tt < NT)
        store_b_run(d, (const v8h*)(HN + ((size_t)bsv[sub] * NT + tt) * NH + g * 32));
      else
        store_b_zero(d);
    }
    __syncthreads();
    __builtin_prefetch(arow + (size_t)(kc + 1) * 256, 0, 3);
#pragma unroll
    for (int s = 0; s < 8; ++s) {
      v16h af = load_a_frag(arow, kc * 256 + s * 32, hl);
#pragma unroll
      for (int sub = 0; sub < 4; ++sub) {
        v16h bf = *(const v16h*)(sB + sub * 4096 + s * 512 + hl * 256 + lm * 16);
        acc[sub] = __builtin_amdgcn_wmma_f32_16x16x32_f16(false, af, false, bf,
                                                          (short)0, acc[sub],
                                                          false, false);
      }
    }
    __syncthreads();
  }

  const int e0 = mBase + hl * 8;
  float biasv[8];
#pragma unroll
  for (int r = 0; r < 8; ++r) biasv[r] = bias[e0 + r];
#pragma unroll
  for (int sub = 0; sub < 4; ++sub) {
    const int n = nBase + sub * 16 + lm;
    const int b = n / NT, t = n - b * NT;
    v8f outv;
#pragma unroll
    for (int r = 0; r < 8; ++r) outv[r] = acc[sub][r] + biasv[r];
    *(v8f*)(zbuf + ((size_t)b * NT + t) * NE + e0) = outv;
  }
}

// ---------------------- per-channel MLP layer 1 (batched GEMM) -------------
// grid (4, 128): m-tiles of h, e. Block tile 64 M x 64 N (all of B). block 128.
__global__ __launch_bounds__(128) void mlp_gemm(const _Float16* __restrict__ P1,
                                                const float* __restrict__ zbuf,
                                                const float* __restrict__ pb1,
                                                float* __restrict__ hmid) {
  __shared__ __align__(32) _Float16 sB[16384];
  const int lane = threadIdx.x & 31;
  const int hl = lane >> 4, lm = lane & 15;
  const int e = blockIdx.y;
  const int mBase = blockIdx.x * 64 + (threadIdx.x >> 5) * 16;   // h rows

  // stage z[b, t, e] fragment-ordered: thread (g, nn) covers t = g*32..g*32+31
  {
    const int g = threadIdx.x >> 4;
    const int nn = threadIdx.x & 15;
    _Float16* dst0 = sB + g * 512 + nn * 16;
#pragma unroll
    for (int sub = 0; sub < 4; ++sub) {
      const int bcol = sub * 16 + nn;
      __align__(32) _Float16 tmp[32];
#pragma unroll
      for (int q = 0; q < 32; ++q) {
        int t = g * 32 + q;
        tmp[q] = (t < NT) ? (_Float16)zbuf[((size_t)bcol * NT + t) * NE + e]
                          : (_Float16)0.0f;
      }
      _Float16* d = dst0 + sub * 4096;
      *(v8h*)(d)       = *(const v8h*)(tmp);
      *(v8h*)(d + 8)   = *(const v8h*)(tmp + 8);
      *(v8h*)(d + 256) = *(const v8h*)(tmp + 16);
      *(v8h*)(d + 264) = *(const v8h*)(tmp + 24);
    }
  }
  __syncthreads();

  v8f acc[4] = {};
  const _Float16* __restrict__ arow = P1 + ((size_t)e * NH + (mBase + lm)) * 256;
#pragma unroll
  for (int s = 0; s < 8; ++s) {
    v16h af = load_a_frag(arow, s * 32, hl);
#pragma unroll
    for (int sub = 0; sub < 4; ++sub) {
      v16h bf = *(const v16h*)(sB + sub * 4096 + s * 512 + hl * 256 + lm * 16);
      acc[sub] = __builtin_amdgcn_wmma_f32_16x16x32_f16(false, af, false, bf,
                                                        (short)0, acc[sub],
                                                        false, false);
    }
  }

  const int h0 = mBase + hl * 8;
  float biasv[8];
#pragma unroll
  for (int r = 0; r < 8; ++r) biasv[r] = pb1[e * NH + h0 + r];
#pragma unroll
  for (int sub = 0; sub < 4; ++sub) {
    const int b = sub * 16 + lm;
    v8f outv;
#pragma unroll
    for (int r = 0; r < 8; ++r) {
      float v = acc[sub][r] + biasv[r];
      outv[r] = v > 0.0f ? v : 0.0f;                 // ReLU fused
    }
    *(v8f*)(hmid + ((size_t)e * NB + b) * NH + h0) = outv;
  }
}

// ---------------------- spectral stats via Parseval ------------------------
__global__ __launch_bounds__(128) void spectral_kernel(const float* __restrict__ zbuf,
                                                       float* __restrict__ amp,
                                                       float* __restrict__ freq,
                                                       float* __restrict__ offs) {
  const int b = blockIdx.x, e = threadIdx.x;
  float s1 = 0.0f, s2 = 0.0f;
  for (int t = 0; t < NT; ++t) {
    float v = zbuf[((size_t)b * NT + t) * NE + e];
    s1 += v;
    s2 += v * v;
  }
  offs[b * NE + e] = s1 * (1.0f / 255.0f);
  if (e >= 1) {
    float sumP = 0.5f * (255.0f * s2 + s1 * s1);     // sum_f |rfft|^2, all 128 bins
    amp[b * 127 + (e - 1)] = 2.0f * sqrtf(sumP) * (1.0f / 255.0f);
    freq[b * 127 + (e - 1)] = (float)e * (1.0f / 255.0f);  // freq_bins[e-1]
  }
}

// ---------------------- MLP layer 2 + atan2 --------------------------------
__global__ __launch_bounds__(256) void phase_kernel(const float* __restrict__ hmid,
                                                    const float* __restrict__ pw2,
                                                    const float* __restrict__ pb2,
                                                    float* __restrict__ phase) {
  const int wave = threadIdx.x >> 5, lane = threadIdx.x & 31;
  const int gid = blockIdx.x * 8 + wave;             // 0..8191
  const int b = gid >> 7, e = gid & 127;
  float a0 = 0.0f, a1 = 0.0f;
  for (int h = lane; h < NH; h += 32) {
    float hm = hmid[((size_t)e * NB + b) * NH + h];
    a0 += hm * pw2[(e * 2 + 0) * NH + h];
    a1 += hm * pw2[(e * 2 + 1) * NH + h];
  }
#pragma unroll
  for (int m = 16; m >= 1; m >>= 1) {
    a0 += __shfl_xor(a0, m, 32);
    a1 += __shfl_xor(a1, m, 32);
  }
  if (lane == 0) {
    a0 += pb2[e * 2 + 0];
    a1 += pb2[e * 2 + 1];
    phase[b * NE + e] = atan2f(a1, a0);
  }
}

// ---------------------------------------------------------------------------
extern "C" void kernel_launch(void* const* d_in, const int* in_sizes, int n_in,
                              void* d_out, int out_size, void* d_ws, size_t ws_size,
                              hipStream_t stream) {
  (void)in_sizes; (void)n_in; (void)out_size; (void)ws_size;
  const float* s       = (const float*)d_in[0];
  const float* conv1_w = (const float*)d_in[1];
  const float* conv1_b = (const float*)d_in[2];
  const float* bn_g    = (const float*)d_in[3];
  const float* bn_b    = (const float*)d_in[4];
  const float* conv2_w = (const float*)d_in[5];
  const float* conv2_b = (const float*)d_in[6];
  const float* pw1     = (const float*)d_in[7];
  const float* pb1     = (const float*)d_in[8];
  const float* pw2     = (const float*)d_in[9];
  const float* pb2     = (const float*)d_in[10];

  char* ws = (char*)d_ws;
  _Float16* A1  = (_Float16*)(ws + OFF_A1);
  _Float16* A2  = (_Float16*)(ws + OFF_A2);
  _Float16* X16 = (_Float16*)(ws + OFF_X16);
  _Float16* P1  = (_Float16*)(ws + OFF_P1);
  float*    H   = (float*)(ws + OFF_H);
  _Float16* HN  = (_Float16*)(ws + OFF_HN);
  float*    Z   = (float*)(ws + OFF_Z);
  float*    HM  = (float*)(ws + OFF_HM);
  float*    ST  = (float*)(ws + OFF_ST);

  float* out   = (float*)d_out;
  float* amp   = out;                 // (64,127)
  float* phase = out + 8128;          // (64,128)
  float* freq  = out + 16320;         // (64,127)
  float* offs  = out + 24448;         // (64,128)

  cast_a1<<<8192, 256, 0, stream>>>(conv1_w, A1, ST);
  cast_a2<<<32768, 256, 0, stream>>>(conv2_w, A2);
  cast_x<<<2040, 256, 0, stream>>>(s, X16);
  cast_p1<<<32768, 256, 0, stream>>>(pw1, P1);

  conv1_gemm<<<dim3(4, 255), 128, 0, stream>>>(A1, X16, conv1_b, H, ST);
  bn_finalize<<<1, 256, 0, stream>>>(bn_g, bn_b, ST);
  bn_apply<<<16320, 256, 0, stream>>>(H, ST, HN);
  conv2_gemm<<<dim3(2, 255), 128, 0, stream>>>(A2, HN, conv2_b, Z);

  mlp_gemm<<<dim3(4, 128), 128, 0, stream>>>(P1, Z, pb1, HM);
  spectral_kernel<<<64, 128, 0, stream>>>(Z, amp, freq, offs);
  phase_kernel<<<1024, 256, 0, stream>>>(HM, pw2, pb2, phase);
}
